// AttrProtoBank_5274219840111
// MI455X (gfx1250) — compile-verified, matrix operand order checked by reading, not run on hardware
//
#include <hip/hip_runtime.h>

// ---------------- problem constants ----------------
#define N_ROWS   16384              // BSZ * SUBTEXT_NUM
#define DIM      512                // PROTO_DIM
#define NPROTO   4096               // PROTO_NUM
#define INV_TEMP (1.0f / 0.9f)

// ---------------- tiling ----------------
#define M_TILE   64                 // rows per block
#define P_CHUNK  32                 // prototypes per chunk (one WMMA K pass)
#define NBLOCKS  (N_ROWS / M_TILE)  // 256
#define NCHUNKS  (NPROTO / P_CHUNK) // 128

// LDS row strides (in bf16 elements), padded so 16B fragment loads with
// row-stride addressing hit distinct banks.
#define ROWA 552                    // sA (64 rows) & sP (32 rows): 1104B rows
#define ROWT 40                     // sPT (512 rows) & sProb (64 rows): 80B rows

// LDS byte layout: sA | buf0{sP,sPT} | buf1{sP,sPT} | sProb | stats
#define SP_BYTES   (32 * ROWA * 2)              // 35328
#define SPT_BYTES  (512 * ROWT * 2)             // 40960
#define BUF_BYTES  (SP_BYTES + SPT_BYTES)       // 76288
#define OFF_BUF    (64 * ROWA * 2)              // 70656 (after sA)
#define OFF_PROB   (OFF_BUF + 2 * BUF_BYTES)    // 223232
#define OFF_STAT   (OFF_PROB + 64 * ROWT * 2)   // 228352
#define SMEM_BYTES (OFF_STAT + 1024 /*rmax,rsum,scale,bestval*/ \
                             + 1024 /*partmax,partsum*/          \
                             + 256  /*bestidx*/ + 512 /*partidx*/ \
                             + 32 /*loss*/ + 16 /*flag*/)        // ~231KB < 320KB/WG

// ---------------- types ----------------
typedef __attribute__((ext_vector_type(16))) __bf16    v16bf;
typedef __attribute__((ext_vector_type(8)))  float     v8f;
typedef __attribute__((ext_vector_type(4)))  float     f32x4;
typedef __attribute__((ext_vector_type(4)))  unsigned  u32x4;
typedef __attribute__((ext_vector_type(2)))  unsigned  u32x2;

struct Frag32B { u32x4 lo, hi; };

__device__ __forceinline__ unsigned short f2bf(float f) {
  // round-to-nearest-even f32 -> bf16 (inputs are finite normals)
  unsigned u = __builtin_bit_cast(unsigned, f);
  return (unsigned short)((u + 0x7FFFu + ((u >> 16) & 1u)) >> 16);
}

__device__ __forceinline__ v16bf load_frag(const unsigned short* p) {
  Frag32B f;
  f.lo = *reinterpret_cast<const u32x4*>(p);        // elems 0..7
  f.hi = *reinterpret_cast<const u32x4*>(p + 16);   // elems 16..23
  return __builtin_bit_cast(v16bf, f);
}

// raw LDS byte offset of a shared-memory pointer (low 32 bits of the
// shared-aperture flat address are the wave-relative LDS address)
__device__ __forceinline__ unsigned lds_off(const void* p) {
  return (unsigned)(unsigned long long)p;
}

// async copy 16B global -> LDS (GVS mode: SGPR64 base + VGPR32 offset),
// tracked by ASYNCcnt
__device__ __forceinline__ void async_g2lds_b128(unsigned lds_byte_off,
                                                 unsigned glb_byte_off,
                                                 const void* base) {
  asm volatile("global_load_async_to_lds_b128 %0, %1, %2"
               :: "v"(lds_byte_off), "v"(glb_byte_off), "s"(base)
               : "memory");
}

__device__ __forceinline__ void wait_async0() {
  asm volatile("s_wait_asynccnt 0x0" ::: "memory");
}

// issue the 64KB async fill of one proto chunk (row-major + transposed)
__device__ __forceinline__ void stage_chunk(unsigned ldsP, unsigned ldsPT,
                                            int p0, int tid,
                                            const void* protoB,
                                            const void* protoT) {
  #pragma unroll
  for (int k = 0; k < 8; ++k) {
    const int i = tid + k * 256;            // 2048 x 16B = 32KB
    const int r = i >> 6, c = i & 63;
    async_g2lds_b128(ldsP + (unsigned)(r * (ROWA * 2) + c * 16),
                     (unsigned)((p0 + r) * (DIM * 2) + c * 16), protoB);
  }
  #pragma unroll
  for (int k = 0; k < 8; ++k) {
    const int i = tid + k * 256;            // 512 rows x 4 x 16B
    const int d = i >> 2, c = i & 3;
    async_g2lds_b128(ldsPT + (unsigned)(d * (ROWT * 2) + c * 16),
                     (unsigned)(d * (NPROTO * 2) + p0 * 2 + c * 16), protoT);
  }
}

// ---------------- prep: proto f32 -> bf16 (row-major) + bf16 transpose ----------------
__global__ void proto_cvt(const float* __restrict__ proto,
                          unsigned short* __restrict__ protoB,
                          unsigned short* __restrict__ protoT) {
  __shared__ unsigned short tile[32][33];
  const int p0 = blockIdx.x * 32, d0 = blockIdx.y * 32;
  const int tx = threadIdx.x, ty = threadIdx.y;     // 32 x 8
  #pragma unroll
  for (int r = 0; r < 4; ++r) {
    const int p = p0 + ty + r * 8;
    const unsigned short b = f2bf(proto[(size_t)p * DIM + d0 + tx]);
    protoB[(size_t)p * DIM + d0 + tx] = b;
    tile[ty + r * 8][tx] = b;
  }
  __syncthreads();
  #pragma unroll
  for (int r = 0; r < 4; ++r) {
    const int d = d0 + ty + r * 8;
    protoT[(size_t)d * NPROTO + p0 + tx] = tile[tx][ty + r * 8];
  }
}

// ---------------- fused flash-softmax codebook kernel ----------------
__global__ void __launch_bounds__(256)
attr_proto_main(const float* __restrict__ flat,
                const float* __restrict__ uni,
                const unsigned short* __restrict__ protoB,
                const unsigned short* __restrict__ protoT,
                float* __restrict__ out,
                float* __restrict__ idxOut,
                float* __restrict__ lossPart) {
  extern __shared__ char smem[];
  unsigned short* sA    = (unsigned short*)smem;               // flatten tile (bf16)
  unsigned short* sPb[2], *sPTb[2];                            // double-buffered chunk
  sPb[0]  = (unsigned short*)(smem + OFF_BUF);
  sPTb[0] = (unsigned short*)(smem + OFF_BUF + SP_BYTES);
  sPb[1]  = (unsigned short*)(smem + OFF_BUF + BUF_BYTES);
  sPTb[1] = (unsigned short*)(smem + OFF_BUF + BUF_BYTES + SP_BYTES);
  unsigned short* sProb = (unsigned short*)(smem + OFF_PROB);  // probs [m][p]
  float* sRmax    = (float*)(smem + OFF_STAT);
  float* sRsum    = sRmax + 64;
  float* sScale   = sRsum + 64;
  float* sBestVal = sScale + 64;
  float* sPartMax = sBestVal + 64;   // [64][2]
  float* sPartSum = sPartMax + 128;  // [64][2]
  int*   sBestIdx = (int*)(sPartSum + 128);
  int*   sPartIdx = sBestIdx + 64;   // [64][2]
  float* sLoss    = (float*)(sPartIdx + 128);
  int*   sFlag    = (int*)(sLoss + 8);

  const int tid  = threadIdx.x;
  const int wave = tid >> 5;
  const int lane = tid & 31;
  const int lh   = lane >> 4;       // which half-wave (K/M split of fragments)
  const int ln   = lane & 15;
  const int row0 = blockIdx.x * M_TILE;
  const int mt1  = wave >> 1;       // GEMM1: m subtile 0..3
  const int pt   = wave & 1;        // GEMM1: p subtile 0..1
  const int d0   = wave * 64;       // GEMM2: this wave's D slice

  const unsigned ldsP[2]  = { lds_off(sPb[0]),  lds_off(sPb[1]) };
  const unsigned ldsPT[2] = { lds_off(sPTb[0]), lds_off(sPTb[1]) };

  if (tid < 64) {
    sRmax[tid] = -1e30f; sRsum[tid] = 0.f;
    sBestVal[tid] = -1e30f; sBestIdx[tid] = 0;
  }

  // kick off async staging of chunk 0 before doing anything else
  stage_chunk(ldsP[0], ldsPT[0], 0, tid, protoB, protoT);

  // stage A tile: 64x512 f32 -> bf16 into LDS (read once per block)
  {
    const f32x4* src = (const f32x4*)(flat + (size_t)row0 * DIM);
    #pragma unroll 4
    for (int k = 0; k < 32; ++k) {
      const int i = tid + k * 256;            // 8192 float4s
      const int r = i >> 7, c = i & 127;
      const f32x4 f = src[i];
      u32x2 pk;
      pk.x = (unsigned)f2bf(f.x) | ((unsigned)f2bf(f.y) << 16);
      pk.y = (unsigned)f2bf(f.z) | ((unsigned)f2bf(f.w) << 16);
      *(u32x2*)(sA + r * ROWA + c * 4) = pk;
    }
  }

  const v8f vzero = {0.f, 0.f, 0.f, 0.f, 0.f, 0.f, 0.f, 0.f};
  v8f acc[4][4];
  #pragma unroll
  for (int a = 0; a < 4; ++a)
    #pragma unroll
    for (int b = 0; b < 4; ++b) acc[a][b] = vzero;

  for (int pc = 0; pc < NCHUNKS; ++pc) {
    const int cur = pc & 1, nxt = cur ^ 1;
    const int p0 = pc * P_CHUNK;

    wait_async0();                  // my fills of buf[cur] done
    __syncthreads();                // everyone's done -> buf[cur] readable;
                                    // also: all reads of buf[nxt] (chunk pc-1) done
    if (tid == 0) *sFlag = 0;
    if (pc + 1 < NCHUNKS)           // overlap next chunk's 64KB fill with compute
      stage_chunk(ldsP[nxt], ldsPT[nxt], p0 + P_CHUNK, tid, protoB, protoT);

    const unsigned short* sP  = sPb[cur];
    const unsigned short* sPT = sPTb[cur];

    // -------- GEMM1: 16x16 dist tile per wave, K = 512 (sw-pipelined) --------
    v8f dist = vzero;
    {
      const unsigned short* arow = sA + (mt1 * 16 + ln) * ROWA + lh * 8;
      const unsigned short* brow = sP + (pt  * 16 + ln) * ROWA + lh * 8;
      v16bf a_cur = load_frag(arow);
      v16bf b_cur = load_frag(brow);
      #pragma unroll
      for (int kc = 0; kc < 16; ++kc) {
        v16bf a_nxt = a_cur, b_nxt = b_cur;
        if (kc < 15) {              // issue next frags before consuming current
          a_nxt = load_frag(arow + (kc + 1) * 32);
          b_nxt = load_frag(brow + (kc + 1) * 32);
        }
        dist = __builtin_amdgcn_wmma_f32_16x16x32_bf16(
            false, a_cur, false, b_cur, (short)0, dist, false, false);
        a_cur = a_nxt; b_cur = b_nxt;
      }
    }

    // -------- gumbel + temperature; chunk row max / argmax --------
    float logit[8];
    const int pcol = p0 + pt * 16 + ln;
    #pragma unroll
    for (int j = 0; j < 8; ++j) {
      const int rloc = mt1 * 16 + j + 8 * lh;
      const float* up = uni + (size_t)(row0 + rloc) * NPROTO + pcol;
      const float u = __builtin_nontemporal_load(up);   // 268MB one-pass stream
      if (pc + 1 < NCHUNKS) __builtin_prefetch(up + P_CHUNK, 0, 0);
      const float g = -__logf(-__logf(u + 1e-10f) + 1e-10f);
      logit[j] = (dist[j] + g) * INV_TEMP;
      float v = logit[j]; int ix = pcol;
      #pragma unroll
      for (int m = 1; m < 16; m <<= 1) {      // reduce over the 16 N-lanes
        const float ov = __shfl_xor(v, m, 32);
        const int   oi = __shfl_xor(ix, m, 32);
        if (ov > v || (ov == v && oi < ix)) { v = ov; ix = oi; }
      }
      if (ln == 0) { sPartMax[rloc * 2 + pt] = v; sPartIdx[rloc * 2 + pt] = ix; }
    }
    __syncthreads();

    if (tid < 64) {                           // merge running max, rescale factor, argmax
      const float m0 = sPartMax[tid * 2], m1 = sPartMax[tid * 2 + 1];
      const float cm = fmaxf(m0, m1);
      const float om = sRmax[tid];
      const float nm = fmaxf(om, cm);
      if (cm > om) {                          // this row's max moved -> rescale needed
        *sFlag = 1;                           // benign race: all writers store 1
        sScale[tid] = __expf(om - nm);
        sRmax[tid]  = nm;
        sRsum[tid] *= sScale[tid];
      } else {
        sScale[tid] = 1.f;
      }
      const int ci = (m0 >= m1) ? sPartIdx[tid * 2] : sPartIdx[tid * 2 + 1];
      if (cm > sBestVal[tid]) { sBestVal[tid] = cm; sBestIdx[tid] = ci; }
    }
    __syncthreads();

    // -------- probs = exp(logit - rowmax) -> bf16 LDS; partial row sums --------
    #pragma unroll
    for (int j = 0; j < 8; ++j) {
      const int rloc = mt1 * 16 + j + 8 * lh;
      const float pv = __expf(logit[j] - sRmax[rloc]);
      sProb[rloc * ROWT + pt * 16 + ln] = f2bf(pv);
      float s = pv;
      #pragma unroll
      for (int m = 1; m < 16; m <<= 1) s += __shfl_xor(s, m, 32);
      if (ln == 0) sPartSum[rloc * 2 + pt] = s;
    }
    __syncthreads();
    if (tid < 64) sRsum[tid] += sPartSum[tid * 2] + sPartSum[tid * 2 + 1];
    const bool needRescale = (*sFlag != 0);   // block-uniform after barrier

    // -------- GEMM2: acc[64 x dslice64] += prob(64x32) @ proto(32 x 64) --------
    v16bf bf[4];
    #pragma unroll
    for (int nt = 0; nt < 4; ++nt)
      bf[nt] = load_frag(sPT + (d0 + nt * 16 + ln) * ROWT + lh * 8);
    v16bf af[4];                              // preload all A frags: hide dscnt
    #pragma unroll
    for (int mt = 0; mt < 4; ++mt)
      af[mt] = load_frag(sProb + (mt * 16 + ln) * ROWT + lh * 8);
    if (needRescale) {                        // rare after warmup; uniform branch
      #pragma unroll
      for (int mt = 0; mt < 4; ++mt) {
        float sc[8];
        #pragma unroll
        for (int j = 0; j < 8; ++j) sc[j] = sScale[mt * 16 + j + 8 * lh];
        #pragma unroll
        for (int nt = 0; nt < 4; ++nt)
          #pragma unroll
          for (int j = 0; j < 8; ++j) acc[mt][nt][j] *= sc[j];
      }
    }
    #pragma unroll
    for (int mt = 0; mt < 4; ++mt)
      #pragma unroll
      for (int nt = 0; nt < 4; ++nt)
        acc[mt][nt] = __builtin_amdgcn_wmma_f32_16x16x32_bf16(
            false, af[mt], false, bf[nt], (short)0, acc[mt][nt], false, false);
  }
  __syncthreads();

  // -------- epilogue: normalize, store output, |out-flat| partials, indices ----
  float lsum = 0.f;
  #pragma unroll
  for (int mt = 0; mt < 4; ++mt) {
    #pragma unroll
    for (int nt = 0; nt < 4; ++nt) {
      #pragma unroll
      for (int j = 0; j < 8; ++j) {
        const int rloc = mt * 16 + j + 8 * lh;
        const size_t gi = (size_t)(row0 + rloc) * DIM + (d0 + nt * 16 + ln);
        const float v = acc[mt][nt][j] / sRsum[rloc];
        __builtin_nontemporal_store(v, out + gi);   // write-once stream
        lsum += fabsf(v - flat[gi]);
      }
    }
  }
  #pragma unroll
  for (int m = 1; m < 32; m <<= 1) lsum += __shfl_xor(lsum, m, 32);
  if (lane == 0) sLoss[wave] = lsum;
  if (tid < 64) idxOut[row0 + tid] = (float)sBestIdx[tid];
  __syncthreads();
  if (tid == 0) {
    float t = 0.f;
    #pragma unroll
    for (int w = 0; w < 8; ++w) t += sLoss[w];
    lossPart[blockIdx.x] = t;
  }
}

// ---------------- deterministic loss reduction (no atomics) ----------------
__global__ void loss_reduce(const float* __restrict__ part,
                            float* __restrict__ lossOut) {
  __shared__ float sh[256];
  sh[threadIdx.x] = part[threadIdx.x];          // NBLOCKS == 256 exactly
  __syncthreads();
  for (int s = 128; s > 0; s >>= 1) {
    if (threadIdx.x < s) sh[threadIdx.x] += sh[threadIdx.x + s];
    __syncthreads();
  }
  if (threadIdx.x == 0) lossOut[0] = sh[0] / (float)((size_t)N_ROWS * DIM);
}

// ---------------- host entry ----------------
extern "C" void kernel_launch(void* const* d_in, const int* in_sizes, int n_in,
                              void* d_out, int out_size, void* d_ws, size_t ws_size,
                              hipStream_t stream) {
  (void)in_sizes; (void)n_in; (void)out_size; (void)ws_size;
  const float* subtexts  = (const float*)d_in[0];
  const float* prototype = (const float*)d_in[1];
  const float* uniform   = (const float*)d_in[2];
  float* out = (float*)d_out;

  // workspace: protoB bf16 (4MB) | protoT bf16 (4MB) | loss partials (1KB)
  unsigned short* protoB = (unsigned short*)d_ws;
  unsigned short* protoT = protoB + (size_t)NPROTO * DIM;
  float* lossPart = (float*)(protoT + (size_t)DIM * NPROTO);

  proto_cvt<<<dim3(NPROTO / 32, DIM / 32), dim3(32, 8), 0, stream>>>(
      prototype, protoB, protoT);
  attr_proto_main<<<NBLOCKS, 256, SMEM_BYTES, stream>>>(
      subtexts, uniform, protoB, protoT,
      out, out + (size_t)N_ROWS * DIM + 1, lossPart);
  loss_reduce<<<1, 256, 0, stream>>>(lossPart, out + (size_t)N_ROWS * DIM);
}
